// cheb_conv_with_Att_GL_26113401159748
// MI455X (gfx1250) — matrix operand order repaired and checked by reading, not compiled, not run on hardware
//
#include <hip/hip_runtime.h>
#include <hip/hip_bf16.h>

#define NN   8
#define TT   64
#define V    200
#define F    64
#define FO   64
#define VP   208   // 13 * 16  (M padding)
#define UP   224   // 7 * 32   (K padding for GEMM-1)
#define J    192   // 3 * 64   (stacked k dimension, = 6*32 exactly)

typedef __attribute__((ext_vector_type(16))) __bf16 v16bf;
typedef __attribute__((ext_vector_type(8)))  float  v8f;

__device__ __forceinline__ unsigned short f2bf(float f) {
  unsigned u = __builtin_bit_cast(unsigned, f);
  u += 0x7FFFu + ((u >> 16) & 1u);           // round-to-nearest-even
  return (unsigned short)(u >> 16);
}

__global__ __launch_bounds__(256, 1)
void cheb_conv_att_kernel(const float* __restrict__ x,
                          const float* __restrict__ Att,
                          const float* __restrict__ S,
                          const float* __restrict__ Theta,
                          float* __restrict__ out) {
  // ---- LDS staging (320,320 B total, within the 320 KB WGP budget) ----
  __shared__ unsigned short a1T[VP * UP];   // A1^T : a1T[v][u] = A1[u,v]      (93,184 B)
  __shared__ unsigned short a2T[VP * UP];   // A2^T                             (93,184 B)
  __shared__ unsigned short xbT[F * UP];    // x^T  : xbT[f][u] = x[u,f]        (28,672 B)
  __shared__ unsigned short rhs[VP * J];    // [rhs0 | rhs1 | rhs2]  (bf16)     (79,872 B)
  __shared__ unsigned short thT[FO * J];    // Theta-stack^T: thT[fo][j]        (24,576 B)
  __shared__ float dcol[VP];                //                                  (832 B)

  const int nt  = blockIdx.x;          // n*T + t
  const int n   = nt / TT;
  const int tid = threadIdx.x;
  const float* Sp = S   + (size_t)nt * V * V;
  const float* xp = x   + (size_t)nt * V * F;
  const float* Ap = Att + (size_t)n  * V * V;

  // ---- Phase A: degree vector d[v] = sum_u min(S[u,v], S[v,u]) (symmetric) ----
  for (int v = tid; v < VP; v += 256) {
    float s = 0.f;
    if (v < V)
      for (int u = 0; u < V; ++u)
        s += fminf(Sp[v * V + u], Sp[u * V + v]);
    dcol[v] = s;
  }

  // ---- Phase A2: stage xbT, thT, rhs0 (independent of d) ----
  for (int i = tid; i < F * UP; i += 256) {
    int f = i / UP, u = i % UP;
    xbT[i] = (u < V) ? f2bf(xp[u * F + f]) : (unsigned short)0;
  }
  for (int i = tid; i < FO * J; i += 256) {
    int fo = i / J, j = i % J;                 // j = k*F + f
    thT[i] = f2bf(Theta[j * FO + fo]);         // Theta[k,f,fo] with stacked leading dims
  }
  for (int i = tid; i < VP * F; i += 256) {    // rhs0[v][f] = Att[v,v] * x[v,f]
    int v = i / F, f = i % F;
    float val = (v < V) ? Ap[v * V + v] * xp[v * F + f] : 0.f;
    rhs[v * J + f] = f2bf(val);
  }
  __syncthreads();

  // ---- Phase B: build A1^T, A2^T in bf16 (L2-resident re-reads of S/Att) ----
  for (int i = tid; i < VP * UP; i += 256) {
    int v = i / UP, u = i % UP;
    unsigned short e1 = 0, e2 = 0;
    if (v < V && u < V) {
      float smin = fminf(Sp[u * V + v], Sp[v * V + u]);
      float att  = Ap[u * V + v];
      float dg   = (u == v) ? 1.f : 0.f;
      float lt   = (dcol[u] - 1.f) * dg - smin;          // L_t[u,v]
      e1 = f2bf(lt * att);                               // A1[u,v]
      e2 = f2bf((2.f * lt * lt - dg) * att);             // A2[u,v]
    }
    a1T[i] = e1;
    a2T[i] = e2;
  }
  __syncthreads();

  const int lane = tid & 31;
  const int wave = tid >> 5;
  const int ln16 = lane & 15;   // M-row (A), N-col (B/D) within a 16x16 tile
  const int h    = lane >> 4;   // lane-half

  // ---- Phase C: GEMM-1  rhs_k[v,f] = sum_u A_k[u,v] * x[u,f]  (k = 1,2) ----
  for (int tile = wave; tile < 104; tile += 8) {      // 104/8 = 13 per wave (uniform)
    int k     = tile / 52;                            // 0 -> A1, 1 -> A2
    int r     = tile % 52;
    int mtile = r >> 2, ntile = r & 3;
    const unsigned short* aT = k ? a2T : a1T;
    const unsigned short* arow = aT  + (mtile * 16 + ln16) * UP;
    const unsigned short* brow = xbT + (ntile * 16 + ln16) * UP;
    v8f acc = {};
#pragma unroll
    for (int kk = 0; kk < UP / 32; ++kk) {
      int ub = kk * 32;
      v16bf af, bf;
      const unsigned short* ap = arow + ub + 8 * h;     // A: k = ub+8h+0..7, +16..+23
      ((uint4*)&af)[0] = *(const uint4*)(ap);
      ((uint4*)&af)[1] = *(const uint4*)(ap + 16);
      const unsigned short* bp = brow + ub + 8 * h;     // B: k = ub+8h+0..7, +16..+23
      ((uint4*)&bf)[0] = *(const uint4*)(bp);
      ((uint4*)&bf)[1] = *(const uint4*)(bp + 16);
      acc = __builtin_amdgcn_wmma_f32_16x16x32_bf16(false, af, false, bf,
                                                    (short)0, acc, false, false);
    }
    int colb = (k + 1) * F + ntile * 16 + ln16;
    int rowb = mtile * 16 + 8 * h;
#pragma unroll
    for (int rr = 0; rr < 8; ++rr)
      rhs[(rowb + rr) * J + colb] = f2bf(acc[rr]);
  }
  __syncthreads();

  // ---- Phase D: GEMM-2  out = relu(rhs(208x192) @ ThetaStack(192x64)) ----
  float* op = out + (size_t)nt * V * FO;
  for (int tile = wave; tile < 52; tile += 8) {       // wave-uniform trip counts
    int mtile = tile >> 2, ntile = tile & 3;
    const unsigned short* arow = rhs + (mtile * 16 + ln16) * J;
    const unsigned short* brow = thT + (ntile * 16 + ln16) * J;
    v8f acc = {};
#pragma unroll
    for (int kk = 0; kk < J / 32; ++kk) {
      int jb = kk * 32;
      v16bf af, bf;
      const unsigned short* ap = arow + jb + 8 * h;
      ((uint4*)&af)[0] = *(const uint4*)(ap);
      ((uint4*)&af)[1] = *(const uint4*)(ap + 16);
      const unsigned short* bp = brow + jb + 8 * h;
      ((uint4*)&bf)[0] = *(const uint4*)(bp);
      ((uint4*)&bf)[1] = *(const uint4*)(bp + 16);
      acc = __builtin_amdgcn_wmma_f32_16x16x32_bf16(false, af, false, bf,
                                                    (short)0, acc, false, false);
    }
    int rowb = mtile * 16 + 8 * h;
    int col  = ntile * 16 + ln16;
#pragma unroll
    for (int rr = 0; rr < 8; ++rr) {
      int vrow = rowb + rr;
      if (vrow < V) {
        float v = acc[rr];
        op[vrow * FO + col] = v > 0.f ? v : 0.f;
      }
    }
  }
}

extern "C" void kernel_launch(void* const* d_in, const int* in_sizes, int n_in,
                              void* d_out, int out_size, void* d_ws, size_t ws_size,
                              hipStream_t stream) {
  const float* x     = (const float*)d_in[0];
  const float* Att   = (const float*)d_in[1];
  const float* S     = (const float*)d_in[2];
  const float* Theta = (const float*)d_in[3];
  float* out = (float*)d_out;
  hipLaunchKernelGGL(cheb_conv_att_kernel, dim3(NN * TT), dim3(256), 0, stream,
                     x, Att, S, Theta, out);
}
